// LoopLinear_21251498180721
// MI455X (gfx1250) — matrix operand order repaired and checked by reading, not compiled, not run on hardware
//
#include <hip/hip_runtime.h>
#include <stdint.h>

typedef __attribute__((ext_vector_type(16))) __bf16 v16bf;
typedef __attribute__((ext_vector_type(8)))  __bf16 v8bf;
typedef __attribute__((ext_vector_type(4)))  __bf16 v4bf;
typedef __attribute__((ext_vector_type(8)))  float  v8f;
typedef __attribute__((ext_vector_type(4)))  float  v4f;

constexpr int KD = 4096;           // IN_DIM  (reduction dim)
constexpr int ND = 4096;           // OUT_DIM
constexpr int MD = 4096;           // batch
constexpr int BM = 128, BN = 128, BK = 32;
constexpr int LDSS = 40;           // padded LDS row stride in bf16 elems (80B, 16B-aligned rows)
constexpr int TILE = 128 * LDSS;   // elems per tile buffer (10KB)
constexpr int NK = KD / BK;        // 128 K-steps

// ---------------------------------------------------------------------------
// Prepass: split f32 -> bf16 (hi) + bf16 (residual lo). Pure bandwidth.
// ---------------------------------------------------------------------------
__global__ __launch_bounds__(256)
void split_bf16_kernel(const float* __restrict__ src,
                       __bf16* __restrict__ hi, __bf16* __restrict__ lo)
{
    const size_t i4 = ((size_t)blockIdx.x * 256 + threadIdx.x) * 4;
    v4f v = *reinterpret_cast<const v4f*>(src + i4);
    v4bf h, l;
#pragma unroll
    for (int i = 0; i < 4; ++i) {
        __bf16 hh = (__bf16)v[i];
        h[i] = hh;
        l[i] = (__bf16)(v[i] - (float)hh);
    }
    *reinterpret_cast<v4bf*>(hi + i4) = h;
    *reinterpret_cast<v4bf*>(lo + i4) = l;
}

// ---------------------------------------------------------------------------
// Main GEMM: bf16x3 split product, async global->LDS staging (ASYNCcnt),
// double-buffered LDS, 24 v_wmma_f32_16x16x32_bf16 per wave per K-step.
// ---------------------------------------------------------------------------
__global__ __launch_bounds__(256)
void gemm_bias_bf16x3_async(const __bf16* __restrict__ Xh, const __bf16* __restrict__ Xl,
                            const __bf16* __restrict__ Wh, const __bf16* __restrict__ Wl,
                            const float* __restrict__ bias, float* __restrict__ out)
{
    // [stage][Ahi, Alo, Bhi, Blo] -> 8 * 10KB = 80KB
    __shared__ __align__(16) unsigned short smem[8 * TILE];

    const int tid   = threadIdx.x;
    const int lane  = tid & 31;          // wave32
    const int wave  = tid >> 5;          // 8 waves
    const int waveM = wave & 3;          // 4 waves along M, 32 rows each
    const int waveN = wave >> 2;         // 2 waves along N, 64 cols each
    const int bm = blockIdx.y * BM;
    const int bn = blockIdx.x * BN;

    // Async copy mapping: one tile = 128 rows x 32 bf16 = 512 x 16B chunks;
    // 256 threads -> 2 chunks per thread per tile, 8 async b128s per stage.
    auto issue_tile = [&](const __bf16* g, int ldsBase, int grow0, int kt) {
#pragma unroll
        for (int q = 0; q < 2; ++q) {
            const int id  = tid + 256 * q;
            const int row = id >> 2;
            const int c   = (id & 3) * 8;      // bf16 elems within the 32-wide row
            const __bf16* gp = g + (size_t)(grow0 + row) * KD + (size_t)kt * BK + c;
            const unsigned lds_off =
                (unsigned)(size_t)(&smem[ldsBase + row * LDSS + c]);   // low 32 bits = LDS byte offset
            asm volatile("global_load_async_to_lds_b128 %0, %1, off"
                         :: "v"(lds_off), "v"((unsigned long long)(size_t)gp)
                         : "memory");
        }
    };

    auto stage_load = [&](int stg, int kt) {
        const int s0 = stg * 4 * TILE;
        issue_tile(Xh, s0,            bm, kt);
        issue_tile(Xl, s0 + TILE,     bm, kt);
        issue_tile(Wh, s0 + 2 * TILE, bn, kt);
        issue_tile(Wl, s0 + 3 * TILE, bn, kt);
    };

    // A 16-bit 16x32 fragment: lanes 0-15 row M=lane hold K 0-15,
    // lanes 16-31 row M=lane-16 hold K 16-31 -> two b128 LDS loads.
    auto ldA = [&](int base, int sub) -> v16bf {
        const int row = waveM * 32 + sub * 16 + (lane & 15);
        const int k0  = (lane < 16) ? 0 : 16;
        const int off = base + row * LDSS + k0;
        union { v16bf v; v8bf h[2]; } u;
        u.h[0] = *reinterpret_cast<const v8bf*>(&smem[off]);
        u.h[1] = *reinterpret_cast<const v8bf*>(&smem[off + 8]);
        return u.v;
    };
    // B 16-bit 32x16 fragment: lane = column N; lanes 0-15 hold K 0-15, lanes 16-31 hold K 16-31.
    auto ldB = [&](int base, int sub) -> v16bf {
        const int row = waveN * 64 + sub * 16 + (lane & 15);
        const int k0  = (lane < 16) ? 0 : 16;
        const int off = base + row * LDSS + k0;
        union { v16bf v; v8bf h[2]; } u;
        u.h[0] = *reinterpret_cast<const v8bf*>(&smem[off]);
        u.h[1] = *reinterpret_cast<const v8bf*>(&smem[off + 8]);
        return u.v;
    };

    const v8f vzero = {0.f, 0.f, 0.f, 0.f, 0.f, 0.f, 0.f, 0.f};
    v8f acc[2][4];
#pragma unroll
    for (int m = 0; m < 2; ++m)
#pragma unroll
        for (int n = 0; n < 4; ++n)
            acc[m][n] = vzero;

    // prologue: stage 0
    stage_load(0, 0);
    asm volatile("s_wait_asynccnt 0x0" ::: "memory");
    __syncthreads();

    for (int kt = 0; kt < NK; ++kt) {
        const int cur = kt & 1;

        if (kt + 1 < NK) stage_load(cur ^ 1, kt + 1);   // async DMA overlaps WMMA below

        const int sA = cur * 4 * TILE;
        const int sB = sA + 2 * TILE;

        v16bf ah[2], al[2];
#pragma unroll
        for (int m = 0; m < 2; ++m) {
            ah[m] = ldA(sA, m);
            al[m] = ldA(sA + TILE, m);
        }

#pragma unroll
        for (int n = 0; n < 4; ++n) {
            v16bf bh = ldB(sB, n);
            v16bf bl = ldB(sB + TILE, n);
#pragma unroll
            for (int m = 0; m < 2; ++m) {
                acc[m][n] = __builtin_amdgcn_wmma_f32_16x16x32_bf16(
                    false, ah[m], false, bh, (short)0, acc[m][n], false, false);
                acc[m][n] = __builtin_amdgcn_wmma_f32_16x16x32_bf16(
                    false, ah[m], false, bl, (short)0, acc[m][n], false, false);
                acc[m][n] = __builtin_amdgcn_wmma_f32_16x16x32_bf16(
                    false, al[m], false, bh, (short)0, acc[m][n], false, false);
            }
        }

        asm volatile("s_wait_asynccnt 0x0" ::: "memory");
        __syncthreads();
    }

    // epilogue: bias + store. C/D layout: VGPR r -> M = r (+8 for lanes 16-31), N = lane&15.
#pragma unroll
    for (int m = 0; m < 2; ++m) {
        const int row0 = bm + waveM * 32 + m * 16 + ((lane >> 4) << 3);
#pragma unroll
        for (int n = 0; n < 4; ++n) {
            const int col = bn + waveN * 64 + n * 16 + (lane & 15);
            const float bv = bias[col];
#pragma unroll
            for (int r = 0; r < 8; ++r) {
                out[(size_t)(row0 + r) * ND + col] = acc[m][n][r] + bv;
            }
        }
    }
}

// ---------------------------------------------------------------------------
// Fallback (round-1 fused kernel): used only if d_ws is too small for the
// bf16 hi/lo staging buffers.
// ---------------------------------------------------------------------------
__global__ __launch_bounds__(256)
void gemm_bias_bf16x3_fused(const float* __restrict__ X,
                            const float* __restrict__ W,
                            const float* __restrict__ bias,
                            float* __restrict__ out)
{
    __shared__ __align__(16) unsigned short smem[8 * TILE];

    const int tid   = threadIdx.x;
    const int lane  = tid & 31;
    const int wave  = tid >> 5;
    const int waveM = wave & 3;
    const int waveN = wave >> 2;
    const int bm = blockIdx.y * BM;
    const int bn = blockIdx.x * BN;

    const float* Xg = X + (size_t)bm * KD;
    const float* Wg = W + (size_t)bn * KD;

    const int rowt = tid >> 3;
    const int kc   = (tid & 7) * 4;

    v4f xr[4], wr[4];

    auto gload = [&](int kt) {
        const size_t kk = (size_t)kt * BK + kc;
#pragma unroll
        for (int q = 0; q < 4; ++q) {
            xr[q] = *reinterpret_cast<const v4f*>(Xg + (size_t)(rowt + 32 * q) * KD + kk);
            wr[q] = *reinterpret_cast<const v4f*>(Wg + (size_t)(rowt + 32 * q) * KD + kk);
        }
    };

    auto cvt_store = [&](int stg) {
        const int sA = stg * 4 * TILE;
        const int sB = sA + 2 * TILE;
#pragma unroll
        for (int q = 0; q < 4; ++q) {
            const int off = (rowt + 32 * q) * LDSS + kc;
            v4bf xh, xl, wh, wl;
#pragma unroll
            for (int i = 0; i < 4; ++i) {
                float xv = xr[q][i];
                __bf16 h = (__bf16)xv;
                xh[i] = h; xl[i] = (__bf16)(xv - (float)h);
                float wv = wr[q][i];
                __bf16 g = (__bf16)wv;
                wh[i] = g; wl[i] = (__bf16)(wv - (float)g);
            }
            *reinterpret_cast<v4bf*>(&smem[sA + off])        = xh;
            *reinterpret_cast<v4bf*>(&smem[sA + TILE + off]) = xl;
            *reinterpret_cast<v4bf*>(&smem[sB + off])        = wh;
            *reinterpret_cast<v4bf*>(&smem[sB + TILE + off]) = wl;
        }
    };

    auto ldA = [&](int base, int sub) -> v16bf {
        const int row = waveM * 32 + sub * 16 + (lane & 15);
        const int k0  = (lane < 16) ? 0 : 16;
        const int off = base + row * LDSS + k0;
        union { v16bf v; v8bf h[2]; } u;
        u.h[0] = *reinterpret_cast<const v8bf*>(&smem[off]);
        u.h[1] = *reinterpret_cast<const v8bf*>(&smem[off + 8]);
        return u.v;
    };
    auto ldB = [&](int base, int sub) -> v16bf {
        const int row = waveN * 64 + sub * 16 + (lane & 15);
        const int k0  = (lane < 16) ? 0 : 16;
        const int off = base + row * LDSS + k0;
        union { v16bf v; v8bf h[2]; } u;
        u.h[0] = *reinterpret_cast<const v8bf*>(&smem[off]);
        u.h[1] = *reinterpret_cast<const v8bf*>(&smem[off + 8]);
        return u.v;
    };

    const v8f vzero = {0.f, 0.f, 0.f, 0.f, 0.f, 0.f, 0.f, 0.f};
    v8f acc[2][4];
#pragma unroll
    for (int m = 0; m < 2; ++m)
#pragma unroll
        for (int n = 0; n < 4; ++n)
            acc[m][n] = vzero;

    gload(0);
    cvt_store(0);
    __syncthreads();

    for (int kt = 0; kt < NK; ++kt) {
        const int cur = kt & 1;
        if (kt + 1 < NK) gload(kt + 1);

        const int sA = cur * 4 * TILE;
        const int sB = sA + 2 * TILE;

        v16bf ah[2], al[2];
#pragma unroll
        for (int m = 0; m < 2; ++m) {
            ah[m] = ldA(sA, m);
            al[m] = ldA(sA + TILE, m);
        }

#pragma unroll
        for (int n = 0; n < 4; ++n) {
            v16bf bh = ldB(sB, n);
            v16bf bl = ldB(sB + TILE, n);
#pragma unroll
            for (int m = 0; m < 2; ++m) {
                acc[m][n] = __builtin_amdgcn_wmma_f32_16x16x32_bf16(
                    false, ah[m], false, bh, (short)0, acc[m][n], false, false);
                acc[m][n] = __builtin_amdgcn_wmma_f32_16x16x32_bf16(
                    false, ah[m], false, bl, (short)0, acc[m][n], false, false);
                acc[m][n] = __builtin_amdgcn_wmma_f32_16x16x32_bf16(
                    false, al[m], false, bh, (short)0, acc[m][n], false, false);
            }
        }

        if (kt + 1 < NK) cvt_store((kt + 1) & 1);
        __syncthreads();
    }

#pragma unroll
    for (int m = 0; m < 2; ++m) {
        const int row0 = bm + waveM * 32 + m * 16 + ((lane >> 4) << 3);
#pragma unroll
        for (int n = 0; n < 4; ++n) {
            const int col = bn + waveN * 64 + n * 16 + (lane & 15);
            const float bv = bias[col];
#pragma unroll
            for (int r = 0; r < 8; ++r) {
                out[(size_t)(row0 + r) * ND + col] = acc[m][n][r] + bv;
            }
        }
    }
}

extern "C" void kernel_launch(void* const* d_in, const int* in_sizes, int n_in,
                              void* d_out, int out_size, void* d_ws, size_t ws_size,
                              hipStream_t stream) {
    const float* x    = (const float*)d_in[0];   // [4096, 4096] row-major (B x IN)
    const float* W    = (const float*)d_in[1];   // [4096, 4096] row-major (OUT x IN)
    const float* bias = (const float*)d_in[2];   // [4096]
    float* out = (float*)d_out;                  // [4096, 4096] (B x OUT)

    const size_t elems = (size_t)KD * ND;                 // 16.7M per matrix
    const size_t need  = 4 * elems * sizeof(unsigned short);  // Xh,Xl,Wh,Wl = 128MB

    dim3 grid(ND / BN, MD / BM);                 // (32, 32)

    if (ws_size >= need) {
        __bf16* Xh = (__bf16*)d_ws;
        __bf16* Xl = Xh + elems;
        __bf16* Wh = Xl + elems;
        __bf16* Wl = Wh + elems;
        const int blocks = (int)(elems / 4 / 256);        // 16384
        split_bf16_kernel<<<blocks, 256, 0, stream>>>(x, Xh, Xl);
        split_bf16_kernel<<<blocks, 256, 0, stream>>>(W, Wh, Wl);
        gemm_bias_bf16x3_async<<<grid, 256, 0, stream>>>(Xh, Xl, Wh, Wl, bias, out);
    } else {
        gemm_bias_bf16x3_fused<<<grid, 256, 0, stream>>>(x, W, bias, out);
    }
}
